// ScatterMoE_83803401879803
// MI455X (gfx1250) — matrix operand order repaired and checked by reading, not compiled, not run on hardware
//
#include <hip/hip_runtime.h>
#include <hip/hip_bf16.h>

// ---------------- problem constants (from reference) ----------------
#define NEXP  8
#define NTOK  8192          // B*S = 4*2048
#define HDIM  1024
#define IDIM  2048

// ---------------- GEMM tiling ----------------
#define BM   128
#define BN   128
#define BK   64             // two WMMA k-steps (K=32 each) per staged tile
#define BSTR 72             // padded column stride (halves) for k-major B tiles

typedef __bf16 v16bf __attribute__((ext_vector_type(16)));
typedef float  v8f   __attribute__((ext_vector_type(8)));
typedef int    v4i   __attribute__((ext_vector_type(4)));

union Frag16 { uint4 q[2]; v16bf v; };

// ---- optional CDNA5 async global->LDS copy (ASYNCcnt path) ----
// Probe-discovered signature: (global int4*, lds int4*, imm offset, imm cpol)
#if defined(__HIP_DEVICE_COMPILE__) && __has_builtin(__builtin_amdgcn_global_load_async_to_lds_b128)
#define HAVE_ASYNC_LDS 1
typedef __attribute__((address_space(1))) v4i g_v4i;
typedef __attribute__((address_space(3))) v4i l_v4i;
#define ASYNC_CP16(gp, lp) \
    __builtin_amdgcn_global_load_async_to_lds_b128((g_v4i*)(gp), (l_v4i*)(lp), 0, 0)
#else
#define HAVE_ASYNC_LDS 0
#define ASYNC_CP16(gp, lp) (*(uint4*)(lp) = *(const uint4*)(gp))
#endif

__device__ __forceinline__ void async_wait0() {
#if HAVE_ASYNC_LDS && __has_builtin(__builtin_amdgcn_s_wait_asynccnt)
    __builtin_amdgcn_s_wait_asynccnt(0);
#endif
}

__device__ __forceinline__ v8f wmma_bf16(v16bf a, v16bf b, v8f c) {
    return __builtin_amdgcn_wmma_f32_16x16x32_bf16(
        /*neg_a=*/false, a, /*neg_b=*/false, b,
        /*c_mod=*/(short)0, c, /*reuse_a=*/false, /*reuse_b=*/false);
}

// A fragment: 16x32 bf16 sub-tile of LDS row-major [row][BK], k-offset kOff.
// Lanes 0-15: K=0..7 then 16..23 ; lanes 16-31: K=8..15 then 24..31 (ISA 7.12.2)
__device__ __forceinline__ v16bf load_a_frag(const __hip_bfloat16* lds, int rowBase, int kOff) {
    int lane = threadIdx.x & 31;
    int r    = rowBase + (lane & 15);
    int k8   = kOff + (lane >> 4) * 8;
    Frag16 f;
    f.q[0] = *(const uint4*)(lds + r * BK + k8);
    f.q[1] = *(const uint4*)(lds + r * BK + 16 + k8);
    return f.v;
}

// B fragment: 32x16 bf16 sub-tile, LDS k-major per column [col][BSTR], k-offset kOff.
// Lane L: col=L&15, K = kOff + (L>>4)*8 + 0..7 then +16 + 0..7 (ISA 7.12.2)
__device__ __forceinline__ v16bf load_b_frag(const __hip_bfloat16* lds, int colBase, int kOff) {
    int lane = threadIdx.x & 31;
    int c    = colBase + (lane & 15);
    int k16  = kOff + (lane >> 4) * 8;
    const uint4* p = (const uint4*)(lds + c * BSTR + k16);
    Frag16 f;
    f.q[0] = p[0];
    f.q[1] = p[2];
    return f.v;
}

// ---------------- kernel 0: zero accumulator + counters ----------------
__global__ void k_zero(float4* __restrict__ outacc, int* __restrict__ counts, long n4) {
    long gid = (long)blockIdx.x * blockDim.x + threadIdx.x;
    if (gid < NEXP) counts[gid] = 0;
    float4 z = {0.f, 0.f, 0.f, 0.f};
    for (long i = gid; i < n4; i += (long)gridDim.x * blockDim.x) outacc[i] = z;
}

// ---------------- kernel 1: router (one wave32 per token) ----------------
__global__ void k_router(const __hip_bfloat16* __restrict__ x,
                         const __hip_bfloat16* __restrict__ rw,
                         __hip_bfloat16* __restrict__ logits_out,
                         int* __restrict__ counts,
                         int* __restrict__ toks,
                         float* __restrict__ gates) {
    int wave = threadIdx.x >> 5;
    int lane = threadIdx.x & 31;
    int tok  = blockIdx.x * (blockDim.x >> 5) + wave;
    if (tok >= NTOK) return;

    const __hip_bfloat16* xr = x + (size_t)tok * HDIM;
    float acc[NEXP] = {};
    for (int h = lane; h < HDIM; h += 32) {
        float xv = __bfloat162float(xr[h]);
#pragma unroll
        for (int e = 0; e < NEXP; ++e)
            acc[e] += xv * __bfloat162float(rw[e * HDIM + h]);
    }
#pragma unroll
    for (int e = 0; e < NEXP; ++e)
#pragma unroll
        for (int off = 16; off > 0; off >>= 1)
            acc[e] += __shfl_xor(acc[e], off, 32);

    if (lane == 0) {
        // raw logits out (bf16, [NTOK, NEXP] after the main output)
#pragma unroll
        for (int e = 0; e < NEXP; ++e)
            logits_out[(size_t)tok * NEXP + e] = __float2bfloat16(acc[e]);

        // softmax
        float mx = acc[0];
#pragma unroll
        for (int e = 1; e < NEXP; ++e) mx = fmaxf(mx, acc[e]);
        float ex[NEXP], sum = 0.0f;
#pragma unroll
        for (int e = 0; e < NEXP; ++e) { ex[e] = __expf(acc[e] - mx); sum += ex[e]; }
        float inv = 1.0f / sum;

        // top-2
        int i0 = 0;
#pragma unroll
        for (int e = 1; e < NEXP; ++e) if (ex[e] > ex[i0]) i0 = e;
        int i1 = (i0 == 0) ? 1 : 0;
#pragma unroll
        for (int e = 0; e < NEXP; ++e) if (e != i0 && ex[e] > ex[i1]) i1 = e;

        float p0 = ex[i0] * inv, p1 = ex[i1] * inv;
        float rs = 1.0f / (p0 + p1);
        p0 *= rs; p1 *= rs;

        int s0 = atomicAdd(&counts[i0], 1);
        toks [i0 * NTOK + s0] = tok;
        gates[i0 * NTOK + s0] = p0;
        int s1 = atomicAdd(&counts[i1], 1);
        toks [i1 * NTOK + s1] = tok;
        gates[i1 * NTOK + s1] = p1;
    }
}

// ---------------- kernel 2: exclusive prefix of expert counts ----------------
__global__ void k_prefix(const int* __restrict__ counts, int* __restrict__ basep) {
    if (threadIdx.x == 0) {
        int b = 0;
        for (int e = 0; e < NEXP; ++e) { basep[e] = b; b += counts[e]; }
    }
}

// ---------------- kernel 3: grouped up-proj + SwiGLU ----------------
// inter[base[e]+slot, i] = silu(x @ w1[e]) * (x @ w2[e])
__global__ __launch_bounds__(256)
void k_up_swiglu(const __hip_bfloat16* __restrict__ x,
                 const __hip_bfloat16* __restrict__ w1,
                 const __hip_bfloat16* __restrict__ w2,
                 const int* __restrict__ counts,
                 const int* __restrict__ basep,
                 const int* __restrict__ toks,
                 __hip_bfloat16* __restrict__ inter) {
    int e       = blockIdx.z;
    int cnt     = counts[e];
    int rowBase = blockIdx.x * BM;
    if (rowBase >= cnt) return;
    int iBase   = blockIdx.y * BN;

    __shared__ __align__(16) __hip_bfloat16 lA [BM * BK];     // row-major [row][BK]
    __shared__ __align__(16) __hip_bfloat16 lB1[BN * BSTR];   // k-major  [col][BSTR]
    __shared__ __align__(16) __hip_bfloat16 lB2[BN * BSTR];
    __shared__ int ltok[BM];

    int t = threadIdx.x;
    if (t < BM) {
        int slot = rowBase + t;
        ltok[t] = (slot < cnt) ? toks[e * NTOK + slot] : 0;
    }
    __syncthreads();

    int lane = t & 31, wave = t >> 5;
    int wm = wave >> 1;       // 0..3  -> rows wm*32
    int wn = wave & 1;        // 0..1  -> cols wn*64

    v8f acc1[8] = {};
    v8f acc2[8] = {};

    const __hip_bfloat16* w1e = w1 + (size_t)e * HDIM * IDIM;
    const __hip_bfloat16* w2e = w2 + (size_t)e * HDIM * IDIM;

    for (int k0 = 0; k0 < HDIM; k0 += BK) {
        // stage A (gathered token rows) via async global->LDS, 16B per lane
#pragma unroll
        for (int c = t; c < (BM * BK) / 8; c += 256) {
            int row = c >> 3, q = (c & 7) * 8;
            ASYNC_CP16(x + (size_t)ltok[row] * HDIM + k0 + q, &lA[row * BK + q]);
        }
        // stage W1/W2: [BK rows of H][BN cols of I] -> LDS k-major [col][BSTR]
#pragma unroll
        for (int c = t; c < (BK * BN) / 8; c += 256) {
            int k  = c >> 4;
            int ic = (c & 15) * 8;
            uint4 v1 = *(const uint4*)(w1e + (size_t)(k0 + k) * IDIM + iBase + ic);
            uint4 v2 = *(const uint4*)(w2e + (size_t)(k0 + k) * IDIM + iBase + ic);
            const __hip_bfloat16* h1p = (const __hip_bfloat16*)&v1;
            const __hip_bfloat16* h2p = (const __hip_bfloat16*)&v2;
#pragma unroll
            for (int j = 0; j < 8; ++j) {
                lB1[(ic + j) * BSTR + k] = h1p[j];
                lB2[(ic + j) * BSTR + k] = h2p[j];
            }
        }
        if (k0 + BK < HDIM) {   // global_prefetch_b8 the next weight tiles
            __builtin_prefetch(w1e + (size_t)(k0 + BK + (t >> 2)) * IDIM + iBase + (t & 3) * 32, 0, 3);
            __builtin_prefetch(w2e + (size_t)(k0 + BK + (t >> 2)) * IDIM + iBase + (t & 3) * 32, 0, 3);
        }
        async_wait0();
        __syncthreads();

#pragma unroll
        for (int ks = 0; ks < BK; ks += 32) {
            v16bf aF[2];
#pragma unroll
            for (int mi = 0; mi < 2; ++mi)
                aF[mi] = load_a_frag(lA, wm * 32 + mi * 16, ks);
#pragma unroll
            for (int ni = 0; ni < 4; ++ni) {
                v16bf b1 = load_b_frag(lB1, wn * 64 + ni * 16, ks);
                v16bf b2 = load_b_frag(lB2, wn * 64 + ni * 16, ks);
#pragma unroll
                for (int mi = 0; mi < 2; ++mi) {
                    acc1[mi * 4 + ni] = wmma_bf16(aF[mi], b1, acc1[mi * 4 + ni]);
                    acc2[mi * 4 + ni] = wmma_bf16(aF[mi], b2, acc2[mi * 4 + ni]);
                }
            }
        }
        __syncthreads();
    }

    // epilogue: silu(h1)*h2 -> bf16 inter (compacted rows)
    int kh  = lane >> 4;
    int col = lane & 15;
    int eb  = basep[e];
#pragma unroll
    for (int mi = 0; mi < 2; ++mi)
#pragma unroll
        for (int ni = 0; ni < 4; ++ni) {
            v8f h1 = acc1[mi * 4 + ni];
            v8f h2 = acc2[mi * 4 + ni];
#pragma unroll
            for (int v = 0; v < 8; ++v) {
                int r    = wm * 32 + mi * 16 + kh * 8 + v;   // M = v + 8*(lane>=16)
                int slot = rowBase + r;
                if (slot < cnt) {
                    float g  = h1[v];
                    float sv = g / (1.0f + __expf(-g));      // silu
                    float val = sv * h2[v];
                    inter[(size_t)(eb + slot) * IDIM + iBase + wn * 64 + ni * 16 + col] =
                        __float2bfloat16(val);
                }
            }
        }
}

// ---------------- kernel 4: grouped down-proj + gated scatter ----------------
// outacc[tok, o] += gate * (inter @ w3[e])
__global__ __launch_bounds__(256)
void k_down(const __hip_bfloat16* __restrict__ inter,
            const __hip_bfloat16* __restrict__ w3,
            const int* __restrict__ counts,
            const int* __restrict__ basep,
            const int* __restrict__ toks,
            const float* __restrict__ gates,
            float* __restrict__ outacc) {
    int e       = blockIdx.z;
    int cnt     = counts[e];
    int rowBase = blockIdx.x * BM;
    if (rowBase >= cnt) return;
    int oBase   = blockIdx.y * BN;

    __shared__ __align__(16) __hip_bfloat16 lA[BM * BK];
    __shared__ __align__(16) __hip_bfloat16 lB[BN * BSTR];
    __shared__ int   ltok[BM];
    __shared__ float lg  [BM];

    int t = threadIdx.x;
    if (t < BM) {
        int slot = rowBase + t;
        ltok[t] = (slot < cnt) ? toks [e * NTOK + slot] : 0;
        lg[t]   = (slot < cnt) ? gates[e * NTOK + slot] : 0.0f;
    }
    __syncthreads();

    int lane = t & 31, wave = t >> 5;
    int wm = wave >> 1;
    int wn = wave & 1;
    int eb = basep[e];

    v8f acc[8] = {};

    const __hip_bfloat16* w3e = w3 + (size_t)e * IDIM * HDIM;

    for (int k0 = 0; k0 < IDIM; k0 += BK) {
        // stage A: compacted inter rows (clamped), async copy
#pragma unroll
        for (int c = t; c < (BM * BK) / 8; c += 256) {
            int row  = c >> 3, q = (c & 7) * 8;
            int slot = rowBase + row;
            int rg   = eb + (slot < cnt ? slot : cnt - 1);
            ASYNC_CP16(inter + (size_t)rg * IDIM + k0 + q, &lA[row * BK + q]);
        }
        // stage W3: [BK rows of I][BN cols of H] -> LDS k-major
#pragma unroll
        for (int c = t; c < (BK * BN) / 8; c += 256) {
            int k  = c >> 4;
            int ic = (c & 15) * 8;
            uint4 v3 = *(const uint4*)(w3e + (size_t)(k0 + k) * HDIM + oBase + ic);
            const __hip_bfloat16* hp = (const __hip_bfloat16*)&v3;
#pragma unroll
            for (int j = 0; j < 8; ++j)
                lB[(ic + j) * BSTR + k] = hp[j];
        }
        if (k0 + BK < IDIM)
            __builtin_prefetch(w3e + (size_t)(k0 + BK + (t >> 2)) * HDIM + oBase + (t & 3) * 32, 0, 3);
        async_wait0();
        __syncthreads();

#pragma unroll
        for (int ks = 0; ks < BK; ks += 32) {
            v16bf aF[2];
#pragma unroll
            for (int mi = 0; mi < 2; ++mi)
                aF[mi] = load_a_frag(lA, wm * 32 + mi * 16, ks);
#pragma unroll
            for (int ni = 0; ni < 4; ++ni) {
                v16bf b = load_b_frag(lB, wn * 64 + ni * 16, ks);
#pragma unroll
                for (int mi = 0; mi < 2; ++mi)
                    acc[mi * 4 + ni] = wmma_bf16(aF[mi], b, acc[mi * 4 + ni]);
            }
        }
        __syncthreads();
    }

    // epilogue: gate * down, atomic scatter into float accumulator
    int kh  = lane >> 4;
    int col = lane & 15;
#pragma unroll
    for (int mi = 0; mi < 2; ++mi)
#pragma unroll
        for (int ni = 0; ni < 4; ++ni) {
            v8f a = acc[mi * 4 + ni];
#pragma unroll
            for (int v = 0; v < 8; ++v) {
                int r    = wm * 32 + mi * 16 + kh * 8 + v;
                int slot = rowBase + r;
                if (slot < cnt) {
                    float val = lg[r] * a[v];
                    atomicAdd(&outacc[(size_t)ltok[r] * HDIM + oBase + wn * 64 + ni * 16 + col], val);
                }
            }
        }
}

// ---------------- kernel 5: float accumulator -> bf16 output (16B stores) ----
__global__ void k_convert(const float4* __restrict__ outacc,
                          uint4* __restrict__ out, long n8) {
    long gid = (long)blockIdx.x * blockDim.x + threadIdx.x;
    for (long i = gid; i < n8; i += (long)gridDim.x * blockDim.x) {
        float4 a = outacc[2 * i];
        float4 b = outacc[2 * i + 1];
        __hip_bfloat16 h[8];
        h[0] = __float2bfloat16(a.x); h[1] = __float2bfloat16(a.y);
        h[2] = __float2bfloat16(a.z); h[3] = __float2bfloat16(a.w);
        h[4] = __float2bfloat16(b.x); h[5] = __float2bfloat16(b.y);
        h[6] = __float2bfloat16(b.z); h[7] = __float2bfloat16(b.w);
        out[i] = *(const uint4*)h;
    }
}

// ---------------- host launcher ----------------
extern "C" void kernel_launch(void* const* d_in, const int* in_sizes, int n_in,
                              void* d_out, int out_size, void* d_ws, size_t ws_size,
                              hipStream_t stream) {
    const __hip_bfloat16* x  = (const __hip_bfloat16*)d_in[0];
    const __hip_bfloat16* rw = (const __hip_bfloat16*)d_in[1];
    const __hip_bfloat16* w1 = (const __hip_bfloat16*)d_in[2];
    const __hip_bfloat16* w2 = (const __hip_bfloat16*)d_in[3];
    const __hip_bfloat16* w3 = (const __hip_bfloat16*)d_in[4];

    __hip_bfloat16* out    = (__hip_bfloat16*)d_out;
    __hip_bfloat16* logits = out + (size_t)NTOK * HDIM;   // output tuple tail

    // workspace layout (~97 MB): counters | base | buckets | gates | f32 accum | bf16 inter
    char*  ws     = (char*)d_ws;
    int*   counts = (int*)ws;
    int*   basep  = (int*)(ws + 256);
    int*   toks   = (int*)(ws + 512);
    size_t off    = 512 + (size_t)NEXP * NTOK * 4;
    float* gates  = (float*)(ws + off);
    off          += (size_t)NEXP * NTOK * 4;
    off           = (off + 1023) & ~(size_t)1023;
    float* outacc = (float*)(ws + off);
    off          += (size_t)NTOK * HDIM * 4;
    __hip_bfloat16* inter = (__hip_bfloat16*)(ws + off);

    k_zero<<<2048, 256, 0, stream>>>((float4*)outacc, counts, (long)NTOK * HDIM / 4);
    k_router<<<NTOK / 8, 256, 0, stream>>>(x, rw, logits, counts, toks, gates);
    k_prefix<<<1, 32, 0, stream>>>(counts, basep);

    dim3 g1(NTOK / BM, IDIM / BN, NEXP);   // (64, 16, 8), early-exit past counts
    k_up_swiglu<<<g1, 256, 0, stream>>>(x, w1, w2, counts, basep, toks, inter);

    dim3 g2(NTOK / BM, HDIM / BN, NEXP);   // (64, 8, 8)
    k_down<<<g2, 256, 0, stream>>>(inter, w3, counts, basep, toks, gates, outacc);

    k_convert<<<4096, 256, 0, stream>>>((const float4*)outacc, (uint4*)out,
                                        (long)NTOK * HDIM / 8);
}
